// BDH_6313601925221
// MI455X (gfx1250) — matrix operand order repaired and checked by reading, not compiled, not run on hardware
//
#include <hip/hip_runtime.h>

// ---------------------------------------------------------------------------
// BDH forward for MI455X (gfx1250): bf16 WMMA GEMMs (v_wmma_f32_16x16x32_bf16),
// wave32, 8 waves/block. BIG config: 128x256 block, 64x64 wave tiles (16 WMMA
// per wave per k-step, 1:1 wmma:ds_load_b128). Async global->LDS staging via
// GLOBAL_LOAD_ASYNC_TO_LDS_B128 when the toolchain exposes the builtin.
// ---------------------------------------------------------------------------

typedef __attribute__((ext_vector_type(16))) __bf16 v16bf;
typedef __attribute__((ext_vector_type(8)))  float  v8f;
typedef int v4i_vs __attribute__((__vector_size__(16)));   // matches builtin param

constexpr int CT  = 1024;   // sequence length
constexpr int CD  = 256;    // model dim
constexpr int CNH = 4;      // heads
constexpr int CN  = 8192;   // sparse dim per head
constexpr int CV  = 256;    // vocab
constexpr int CLAYERS = 6;

enum { EPI_F32 = 0, EPI_RELU_F32 = 1, EPI_MASK_BF16 = 2, EPI_RELU_MUL_BF16 = 3 };

#if __has_builtin(__builtin_amdgcn_global_load_async_to_lds_b128) && \
    __has_builtin(__builtin_amdgcn_s_wait_asynccnt)
#define USE_ASYNC_LDS 1
#endif

// Copy 16B global -> LDS. Async path: GLOBAL_LOAD_ASYNC_TO_LDS_B128 (ASYNCcnt),
// no VGPR bounce. Pointers passed directly (provenance intact) so the backend's
// InferAddressSpaces recovers AS1 (global) / AS3 (LDS).
__device__ __forceinline__ void stage16(const __bf16* g, __bf16* l) {
#ifdef USE_ASYNC_LDS
  __builtin_amdgcn_global_load_async_to_lds_b128((v4i_vs*)g, (v4i_vs*)l, 0, 0);
#else
  *(uint4*)l = *(const uint4*)g;
#endif
}

__device__ __forceinline__ void stage_wait() {
#ifdef USE_ASYNC_LDS
  __builtin_amdgcn_s_wait_asynccnt(0);
#endif
}

// ---------------------------------------------------------------------------
// Generic GEMM: C[m][n] = sum_k A[m*lda+k] * Bt[n*ldbt+k]  (A,Bt bf16, acc f32)
// BIG:  block 128(M) x 256(N), 8 waves as 2x4, each wave 64x64 = 4x4 WMMA frags
// small:block 128(M) x  64(N), 8 waves as 4x2, each wave 32x32 = 2x2 WMMA frags
// ---------------------------------------------------------------------------
template <int EPI, bool BIG>
__global__ __launch_bounds__(256) void k_gemm(
    const __bf16* __restrict__ A,  int lda,  long long sAz,
    const __bf16* __restrict__ Bt, int ldbt, long long sBz,
    float* __restrict__ outF, __bf16* __restrict__ outH, long long sOz, int ldo,
    const float* __restrict__ aux, long long sXz,
    int K)
{
  constexpr int BM = 128;
  constexpr int BN = BIG ? 256 : 64;
  constexpr int FM = BIG ? 4 : 2;         // 16-row fragments per wave (M)
  constexpr int FN = BIG ? 4 : 2;         // 16-col fragments per wave (N)
  constexpr int ACH = (BM * 4) / 256;     // A b128 chunks per thread
  constexpr int BCH = (BN * 4) / 256;     // B b128 chunks per thread

  const int z    = blockIdx.z;
  const int m0   = blockIdx.y * BM;
  const int n0   = blockIdx.x * BN;
  const int tid  = threadIdx.x;
  const int lane = tid & 31;
  const int wave = tid >> 5;
  const int wm   = BIG ? ((wave & 1) << 6) : ((wave & 3) << 5);
  const int wn   = BIG ? ((wave >> 1) << 6) : ((wave >> 2) << 5);
  const int lrow = lane & 15;
  const int half = lane >> 4;

  A  += (long long)z * sAz;
  Bt += (long long)z * sBz;

  __shared__ __attribute__((aligned(16))) __bf16 sA[BM * 40];
  __shared__ __attribute__((aligned(16))) __bf16 sB[BN * 40];

  v8f acc[FM][FN] = {};

  // Strictly-causal scores: tile with n0 >= m0+128 is entirely masked.
  const bool fullyMasked = (EPI == EPI_MASK_BF16) && (n0 >= m0 + BM);

  if (!fullyMasked) {
    for (int k0 = 0; k0 < K; k0 += 32) {
      // Stage A tile (BM x 32 bf16) and Bt tile (BN x 32 bf16).
#pragma unroll
      for (int i = 0; i < ACH; ++i) {
        int c = tid + (i << 8);
        int m = c >> 2, kc = (c & 3) << 3;
        stage16(&A[(long long)(m0 + m) * lda + (k0 + kc)], &sA[m * 40 + kc]);
      }
#pragma unroll
      for (int i = 0; i < BCH; ++i) {
        int c = tid + (i << 8);
        int n = c >> 2, kc = (c & 3) << 3;
        stage16(&Bt[(long long)(n0 + n) * ldbt + (k0 + kc)], &sB[n * 40 + kc]);
      }
      stage_wait();
      __syncthreads();

      union Frag { v16bf v; unsigned u[8]; };
      // B 32x16 frags (this code's contiguous k order, same as A): VGPR v -> K = half*16 + v*2 (k contiguous
      // in n-major LDS tile -> coalesces to 2 x ds_load_b128 per fragment).
      Frag b[FN];
#pragma unroll
      for (int f = 0; f < FN; ++f) {
        const __bf16* pb = &sB[(wn + (f << 4) + lrow) * 40];
#pragma unroll
        for (int vv = 0; vv < 8; ++vv)
          b[f].u[vv] = *(const unsigned*)(pb + (half << 4) + (vv << 1));
      }
      // A 16x32 frags: VGPR v -> K = half*16 + v*2.
#pragma unroll
      for (int fm = 0; fm < FM; ++fm) {
        Frag a;
        const __bf16* pa = &sA[(wm + (fm << 4) + lrow) * 40];
#pragma unroll
        for (int vv = 0; vv < 8; ++vv)
          a.u[vv] = *(const unsigned*)(pa + (half << 4) + (vv << 1));
#pragma unroll
        for (int fn = 0; fn < FN; ++fn)
          acc[fm][fn] = __builtin_amdgcn_wmma_f32_16x16x32_bf16(
              false, a.v, false, b[fn].v, (short)0, acc[fm][fn], false, false);
      }
      __syncthreads();
    }
  }

  // Epilogue. C layout: VGPR r -> row = half*8 + r, col = lane&15.
#pragma unroll
  for (int fm = 0; fm < FM; ++fm) {
#pragma unroll
    for (int fn = 0; fn < FN; ++fn) {
#pragma unroll
      for (int r = 0; r < 8; ++r) {
        int row = m0 + wm + (fm << 4) + (half << 3) + r;
        int col = n0 + wn + (fn << 4) + lrow;
        float v = acc[fm][fn][r];
        long long o = (long long)z * sOz + (long long)row * ldo + col;
        if (EPI == EPI_F32) {
          outF[o] = v;
        } else if (EPI == EPI_RELU_F32) {
          outF[o] = fmaxf(v, 0.f);
        } else if (EPI == EPI_MASK_BF16) {
          outH[o] = (__bf16)((col < row) ? v : 0.f);
        } else if (EPI == EPI_RELU_MUL_BF16) {
          float xv = aux[(long long)z * sXz + (long long)row * ldo + col];
          outH[o] = (__bf16)(fmaxf(v, 0.f) * xv);
        }
      }
    }
  }
}

// ---------------------------------------------------------------------------
// Elementwise / reduction kernels
// ---------------------------------------------------------------------------
__device__ inline float blockSum256(float v, float* red) {
  int d = threadIdx.x;
  red[d] = v;
  __syncthreads();
  for (int s = 128; s > 0; s >>= 1) {
    if (d < s) red[d] += red[d + s];
    __syncthreads();
  }
  float r = red[0];
  __syncthreads();
  return r;
}

// rope tables: cos/sin over (T, N); THETA = 2^16 -> freq = exp2(-16*q/N)/(2pi)
__global__ void k_rope_table(float* __restrict__ cosT, float* __restrict__ sinT) {
  long long i = (long long)blockIdx.x * 256 + threadIdx.x;  // over T*N
  int t = (int)(i / CN);
  int n = (int)(i % CN);
  float q = (float)(n & ~1);
  float freq = exp2f(-16.f * q / (float)CN) * 0.15915494309189535f;
  float ph = fmodf((float)t * freq, 1.f) * 6.283185307179586f;
  cosT[i] = cosf(ph);
  sinT[i] = sinf(ph);
}

// QR = rope(x_sparse) -> bf16 ; interleaved pair rotation
__global__ void k_rope(const float* __restrict__ xsp, const float* __restrict__ cosT,
                       const float* __restrict__ sinT, __bf16* __restrict__ qr) {
  long long g = (long long)blockIdx.x * 256 + threadIdx.x;  // pair index over NH*T*N/2
  long long tn = g % ((long long)CT * (CN / 2));
  int t = (int)(tn / (CN / 2));
  int p = (int)(tn % (CN / 2));
  float2 v = ((const float2*)xsp)[g];
  long long ci = (long long)t * (CN / 2) + p;
  float2 c = ((const float2*)cosT)[ci];
  float2 s = ((const float2*)sinT)[ci];
  qr[2 * g + 0] = (__bf16)(v.x * c.x - v.y * s.x);
  qr[2 * g + 1] = (__bf16)(v.y * c.y + v.x * s.y);
}

// transpose + f32->bf16: dst[z][c][r] = src[z][r][c]
__global__ void k_transpose_bf16(const float* __restrict__ src, __bf16* __restrict__ dst,
                                 int R, int C, long long srcZ, long long dstZ) {
  long long i = (long long)blockIdx.x * 256 + threadIdx.x;
  int z = blockIdx.y;
  if (i >= (long long)R * C) return;
  int r = (int)(i / C), c = (int)(i % C);
  dst[(long long)z * dstZ + (long long)c * R + r] = (__bf16)src[(long long)z * srcZ + i];
}

// x = layernorm(embed[idx]); also bf16 row-major + bf16 transposed copies
__global__ void k_embed_ln(const int* __restrict__ idx, const float* __restrict__ embed,
                           float* __restrict__ x, __bf16* __restrict__ xH,
                           __bf16* __restrict__ xT) {
  __shared__ float red[256];
  int t = blockIdx.x, d = threadIdx.x;
  float v = embed[(long long)idx[t] * CD + d];
  float m = blockSum256(v, red) * (1.f / CD);
  float dv = v - m;
  float var = blockSum256(dv * dv, red) * (1.f / CD);
  float y = dv * rsqrtf(var + 1e-5f);
  x[t * CD + d] = y;
  xH[t * CD + d] = (__bf16)y;
  xT[(long long)d * CT + t] = (__bf16)y;
}

// per-row layernorm (row length 256) -> bf16
__global__ void k_ln_rows_bf16(const float* __restrict__ src, __bf16* __restrict__ dst) {
  __shared__ float red[256];
  long long row = blockIdx.x;
  int d = threadIdx.x;
  float v = src[row * CD + d];
  float m = blockSum256(v, red) * (1.f / CD);
  float dv = v - m;
  float var = blockSum256(dv * dv, red) * (1.f / CD);
  dst[row * CD + d] = (__bf16)(dv * rsqrtf(var + 1e-5f));
}

// x = LN(x + LN(sum of 16 split-K partials)); refresh x copies (f32/bf16/bf16-T)
__global__ void k_resid_ln(const float* __restrict__ yMLPp, float* __restrict__ x,
                           __bf16* __restrict__ xH, __bf16* __restrict__ xT) {
  __shared__ float red[256];
  int t = blockIdx.x, d = threadIdx.x;
  float v = 0.f;
#pragma unroll
  for (int j = 0; j < 16; ++j) v += yMLPp[(long long)j * CT * CD + t * CD + d];
  float m1 = blockSum256(v, red) * (1.f / CD);
  float dv = v - m1;
  float var1 = blockSum256(dv * dv, red) * (1.f / CD);
  float yln = dv * rsqrtf(var1 + 1e-5f);
  float u = x[t * CD + d] + yln;
  float m2 = blockSum256(u, red) * (1.f / CD);
  float du = u - m2;
  float var2 = blockSum256(du * du, red) * (1.f / CD);
  float xn = du * rsqrtf(var2 + 1e-5f);
  x[t * CD + d] = xn;
  xH[t * CD + d] = (__bf16)xn;
  xT[(long long)d * CT + t] = (__bf16)xn;
}

// ---------------------------------------------------------------------------
// Launch
// ---------------------------------------------------------------------------
extern "C" void kernel_launch(void* const* d_in, const int* in_sizes, int n_in,
                              void* d_out, int out_size, void* d_ws, size_t ws_size,
                              hipStream_t stream) {
  (void)in_sizes; (void)n_in; (void)out_size; (void)ws_size;
  const int*   idx   = (const int*)d_in[0];
  const float* embed = (const float*)d_in[1];
  const float* enc   = (const float*)d_in[2];
  const float* encv  = (const float*)d_in[3];
  const float* dec   = (const float*)d_in[4];
  const float* lmh   = (const float*)d_in[5];
  float* logits = (float*)d_out;

  size_t off = 0;
  auto carve = [&](size_t bytes) -> char* {
    char* q = (char*)d_ws + off;
    off = (off + bytes + 255) & ~(size_t)255;
    return q;
  };
  float*  cosT  = (float*)carve((size_t)CT * CN * 4);
  float*  sinT  = (float*)carve((size_t)CT * CN * 4);
  __bf16* encT  = (__bf16*)carve((size_t)CNH * CN * CD * 2);  // (h, N, D)
  __bf16* encvT = (__bf16*)carve((size_t)CNH * CN * CD * 2);  // (h, N, D)
  __bf16* decT  = (__bf16*)carve((size_t)CD * CNH * CN * 2);  // (D, NH*N)
  __bf16* lmT   = (__bf16*)carve((size_t)CV * CD * 2);        // (V, D)
  float*  x     = (float*)carve((size_t)CT * CD * 4);
  __bf16* xH    = (__bf16*)carve((size_t)CT * CD * 2);
  __bf16* xTH   = (__bf16*)carve((size_t)CD * CT * 2);
  float*  xsp   = (float*)carve((size_t)CNH * CT * CN * 4);
  __bf16* qrH   = (__bf16*)carve((size_t)CNH * CT * CN * 2);  // reused as xy (bf16)
  __bf16* scH   = (__bf16*)carve((size_t)CNH * CT * CT * 2);
  float*  yKV   = (float*)carve((size_t)CNH * CT * CD * 4);
  __bf16* yKVH  = (__bf16*)carve((size_t)CNH * CT * CD * 2);
  float*  yMLPp = (float*)carve((size_t)16 * CT * CD * 4);    // split-K partials

  // --- one-time prep (recomputed every call: deterministic) ---
  k_rope_table<<<dim3(((long long)CT * CN) / 256), 256, 0, stream>>>(cosT, sinT);
  k_transpose_bf16<<<dim3((CD * CN) / 256, CNH), 256, 0, stream>>>(
      enc, encT, CD, CN, (long long)CD * CN, (long long)CN * CD);
  k_transpose_bf16<<<dim3((CD * CN) / 256, CNH), 256, 0, stream>>>(
      encv, encvT, CD, CN, (long long)CD * CN, (long long)CN * CD);
  k_transpose_bf16<<<dim3((CNH * CN * CD) / 256, 1), 256, 0, stream>>>(
      dec, decT, CNH * CN, CD, 0, 0);
  k_transpose_bf16<<<dim3((CD * CV) / 256, 1), 256, 0, stream>>>(lmh, lmT, CD, CV, 0, 0);
  k_embed_ln<<<CT, 256, 0, stream>>>(idx, embed, x, xH, xTH);

  for (int L = 0; L < CLAYERS; ++L) {
    // x_sparse = relu(x @ enc[h])            (M=T, N=CN, K=D)  [BIG]
    k_gemm<EPI_RELU_F32, true><<<dim3(CN / 256, CT / 128, CNH), 256, 0, stream>>>(
        xH, CD, 0ll, encT, CD, (long long)CN * CD,
        xsp, nullptr, (long long)CT * CN, CN, nullptr, 0ll, CD);
    // QR = rope(x_sparse) -> bf16
    k_rope<<<dim3(((long long)CNH * CT * (CN / 2)) / 256), 256, 0, stream>>>(
        xsp, cosT, sinT, qrH);
    // scores = (QR @ QR^T) * strict-causal   (M=T, N=T, K=CN)  [BIG + tile skip]
    k_gemm<EPI_MASK_BF16, true><<<dim3(CT / 256, CT / 128, CNH), 256, 0, stream>>>(
        qrH, CN, (long long)CT * CN, qrH, CN, (long long)CT * CN,
        nullptr, scH, (long long)CT * CT, CT, nullptr, 0ll, CN);
    // yKV = scores @ x                       (M=T, N=D, K=T)   [small]
    k_gemm<EPI_F32, false><<<dim3(CD / 64, CT / 128, CNH), 256, 0, stream>>>(
        scH, CT, (long long)CT * CT, xTH, CT, 0ll,
        yKV, nullptr, (long long)CT * CD, CD, nullptr, 0ll, CT);
    // yKV = layernorm(yKV) -> bf16
    k_ln_rows_bf16<<<CNH * CT, 256, 0, stream>>>(yKV, yKVH);
    // xy = relu(yKV @ enc_v[h]) * x_sparse -> bf16 (reuse qrH) [BIG]
    k_gemm<EPI_RELU_MUL_BF16, true><<<dim3(CN / 256, CT / 128, CNH), 256, 0, stream>>>(
        yKVH, CD, (long long)CT * CD, encvT, CD, (long long)CN * CD,
        nullptr, qrH, (long long)CT * CN, CN, xsp, (long long)CT * CN, CD);
    // yMLP partials: per head, deterministic split-K over 4 slices of K=8192.
    // Partial j = h*4 + z holds xy[h][:, z*2048:(z+1)*2048] @ dec-slice.
    for (int h = 0; h < CNH; ++h) {
      k_gemm<EPI_F32, false><<<dim3(CD / 64, CT / 128, 4), 256, 0, stream>>>(
          qrH + (long long)h * CT * CN, CN, 2048ll,
          decT + (long long)h * CN, CNH * CN, 2048ll,
          yMLPp + (long long)h * 4 * CT * CD, nullptr, (long long)CT * CD, CD,
          nullptr, 0ll, 2048);
    }
    // x = LN(x + LN(sum partials))
    k_resid_ln<<<CT, 256, 0, stream>>>(yMLPp, x, xH, xTH);
  }

  // logits = x @ lm_head                      (M=T, N=V, K=D)  [small]
  k_gemm<EPI_F32, false><<<dim3(CV / 64, CT / 128, 1), 256, 0, stream>>>(
      xH, CD, 0ll, lmT, CD, 0ll,
      logits, nullptr, 0ll, CV, nullptr, 0ll, CD);
}
